// Graphormer3D_63513976374007
// MI455X (gfx1250) — matrix operand order repaired and checked by reading, not compiled, not run on hardware
//
#include <hip/hip_runtime.h>

// ---------------------------------------------------------------------------
// Graphormer3D forward for MI455X (gfx1250), all GEMMs on bf16 WMMA
// B=8, N=256, D=768, H=12, K=128, L=4, BLOCKS=2, HD=64
// Vectorized staging (b128 global loads, b64 LDS stores, b128 fragment loads),
// templated on B-operand transpose with strength-reduced pointer stepping.
// ---------------------------------------------------------------------------

#define Bc   8
#define Nc   256
#define Dc   768
#define Hc   12
#define Kc   128
#define HDc  64
#define ROWS (Nc * Bc)          // 2048 token rows, row = i*B + b  ([N,B,D])
#define NEGB (-1e9f)

typedef __attribute__((ext_vector_type(16))) __bf16 v16bf;
typedef __attribute__((ext_vector_type(8)))  __bf16 v8bf;
typedef __attribute__((ext_vector_type(4)))  __bf16 v4bf;
typedef __attribute__((ext_vector_type(2)))  __bf16 v2bf;
typedef __attribute__((ext_vector_type(8)))  float  v8f;

__device__ __forceinline__ v8f zero8() {
  v8f z = {0.f,0.f,0.f,0.f,0.f,0.f,0.f,0.f};
  return z;
}
__device__ __forceinline__ v4bf pack4(float4 v) {
  v4bf r; r[0] = (__bf16)v.x; r[1] = (__bf16)v.y; r[2] = (__bf16)v.z; r[3] = (__bf16)v.w;
  return r;
}
__device__ __forceinline__ v16bf cat16(v8bf lo, v8bf hi) {
  return __builtin_shufflevector(lo, hi, 0,1,2,3,4,5,6,7,8,9,10,11,12,13,14,15);
}
__device__ __forceinline__ float gelu_exact(float v) {
  return 0.5f * v * (1.0f + erff(v * 0.70710678118f));
}

// ---------------------------------------------------------------------------
// Generic batched GEMM: C = act(alpha * A@B + bias) + R
//   A: (m,k) = A[m*lda + k]            (after batch offset)
//   B: OPB=0: (k,n) = B[k*ldb + n];  OPB=1: (k,n) = B[n*ldb + k]
//   batch z: z1 = z/zdiv, z2 = z%zdiv; ptr += z1*s?1 + z2*s?2
// Requires M%64==0 (gridDim.y=M/64), N%64==0 (gridDim.x=N/64), K%32==0,
// lda/ldb multiples of 4 (float4 loads).
// LDS tiles K-minor (B transposed) -> each WMMA fragment is two 16-byte
// ds_load_b128's per lane (CDNA5 16-bit A/B lane layouts, ISA 7.12.2).
// ---------------------------------------------------------------------------
template <int OPB>
__global__ __launch_bounds__(256) void gemm_bf16_wmma(
    const float* __restrict__ A,  long lda, long sA1, long sA2,
    const float* __restrict__ Bm, long ldb, long sB1, long sB2,
    const float* __restrict__ bias,
    const float* __restrict__ R,  long ldr, long sR1, long sR2,
    float* __restrict__ C,        long ldc, long sC1, long sC2,
    int zdiv, int K, float alpha, int act_gelu)
{
  __shared__ __bf16 As[64][40];   // [m][k], row = 80B (16B aligned)
  __shared__ __bf16 Bs[64][40];   // [n][k], transposed

  const int z  = blockIdx.z;
  const long z1 = z / zdiv, z2 = z % zdiv;
  A  += z1 * sA1 + z2 * sA2;
  Bm += z1 * sB1 + z2 * sB2;
  if (R) R += z1 * sR1 + z2 * sR2;
  C  += z1 * sC1 + z2 * sC2;

  const int tid  = threadIdx.x;
  const int wave = tid >> 5, lane = tid & 31;
  const int wm = wave & 3, wn = wave >> 2;          // 4x2 waves -> 64x64 tile
  const int m_blk = blockIdx.y * 64;
  const int n_blk = blockIdx.x * 64;

  // staging coordinates (fixed per thread)
  const int ar  = tid >> 3;            // 0..31  (A rows, +32 for 2nd item)
  const int ac4 = (tid & 7) * 4;       // A k-offset
  const int bkr = tid >> 4;            // 0..15  (OPB=0: B k-row, +16)
  const int bn4 = (tid & 15) * 4;      // OPB=0: B n-offset

  const int mrow = wm * 16 + (lane & 15);
  const int hiA  = (lane >= 16) ? 8 : 0;
  const int hiB  = (lane >= 16) ? 8 : 0;
  const int nc   = wn * 32 + (lane & 15);

  // strength-reduced global pointers, stepped each K-iteration
  const float* pa0 = A + (long)(m_blk + ar) * lda + ac4;
  const float* pa1 = pa0 + 32 * lda;
  const float* pb0;
  const float* pb1;
  long bstep;
  if (OPB) {
    pb0 = Bm + (long)(n_blk + ar) * ldb + ac4;
    pb1 = pb0 + 32 * ldb;
    bstep = 32;
  } else {
    pb0 = Bm + (long)bkr * ldb + n_blk + bn4;
    pb1 = pb0 + 16 * ldb;
    bstep = 32 * ldb;
  }

  v8f acc0 = zero8(), acc1 = zero8();

  for (int k0 = 0; k0 < K; k0 += 32) {
    // Issue all global loads first (overlap the barrier wait below).
    float4 a0 = *(const float4*)pa0;
    float4 a1 = *(const float4*)pa1;
    float4 b0 = *(const float4*)pb0;
    float4 b1 = *(const float4*)pb1;
    if (k0 + 32 < K)
      __builtin_prefetch(pa0 + 32, 0, 3);
    pa0 += 32; pa1 += 32; pb0 += bstep; pb1 += bstep;
    __syncthreads();                   // previous tile fully consumed

    *(v4bf*)&As[ar][ac4]      = pack4(a0);
    *(v4bf*)&As[ar + 32][ac4] = pack4(a1);
    if (OPB) {
      *(v4bf*)&Bs[ar][ac4]      = pack4(b0);
      *(v4bf*)&Bs[ar + 32][ac4] = pack4(b1);
    } else {                           // transpose on the way in
      Bs[bn4+0][bkr] = (__bf16)b0.x; Bs[bn4+1][bkr] = (__bf16)b0.y;
      Bs[bn4+2][bkr] = (__bf16)b0.z; Bs[bn4+3][bkr] = (__bf16)b0.w;
      Bs[bn4+0][bkr+16] = (__bf16)b1.x; Bs[bn4+1][bkr+16] = (__bf16)b1.y;
      Bs[bn4+2][bkr+16] = (__bf16)b1.z; Bs[bn4+3][bkr+16] = (__bf16)b1.w;
    }
    __syncthreads();

    v16bf af  = cat16(*(const v8bf*)&As[mrow][hiA], *(const v8bf*)&As[mrow][16 + hiA]);
    v16bf bf0 = cat16(*(const v8bf*)&Bs[nc][hiB],      *(const v8bf*)&Bs[nc][16 + hiB]);
    v16bf bf1 = cat16(*(const v8bf*)&Bs[nc + 16][hiB], *(const v8bf*)&Bs[nc + 16][16 + hiB]);

    acc0 = __builtin_amdgcn_wmma_f32_16x16x32_bf16(false, af, false, bf0,
                                                   (short)0, acc0, false, false);
    acc1 = __builtin_amdgcn_wmma_f32_16x16x32_bf16(false, af, false, bf1,
                                                   (short)0, acc1, false, false);
  }

  // epilogue: C/D layout -> m = g + 8*(lane>=16), n = lane&15
  const int mo = m_blk + wm * 16 + ((lane >= 16) ? 8 : 0);
  const int no = n_blk + wn * 32 + (lane & 15);
  #pragma unroll
  for (int g = 0; g < 8; ++g) {
    #pragma unroll
    for (int t = 0; t < 2; ++t) {
      float v = (t == 0 ? acc0[g] : acc1[g]) * alpha;
      int n = no + t * 16;
      long m = mo + g;
      if (bias) v += bias[n];
      if (act_gelu) v = gelu_exact(v);
      if (R) v += R[m * ldr + n];
      C[m * ldc + n] = v;
    }
  }
}

// ---------------------------------------------------------------------------
// ef[b,n,k] = sum_j !pad_j * gauss(mul[et]*dist+bias[et]; mean_k, std_k)
// ---------------------------------------------------------------------------
__global__ __launch_bounds__(128) void ef_kernel(
    const int* __restrict__ atoms, const float* __restrict__ pos,
    const float* __restrict__ means, const float* __restrict__ stds,
    const float* __restrict__ mul,  const float* __restrict__ gbias,
    float* __restrict__ ef)
{
  __shared__ float e_s[Nc];
  __shared__ int   pad_s[Nc];
  const int row = blockIdx.x;          // b*N + n
  const int b = row / Nc;
  const int t = threadIdx.x;
  const float px = pos[row*3+0], py = pos[row*3+1], pz = pos[row*3+2];
  const int ai = atoms[row];
  for (int j = t; j < Nc; j += 128) {
    int aj = atoms[b*Nc + j];
    float dx = px - pos[(b*Nc+j)*3+0];
    float dy = py - pos[(b*Nc+j)*3+1];
    float dz = pz - pos[(b*Nc+j)*3+2];
    float sq = dx*dx + dy*dy + dz*dz;
    float dist = (sq > 0.f) ? sqrtf(sq) : 0.f;
    int et = ai*10 + aj;
    e_s[j]   = mul[et]*dist + gbias[et];
    pad_s[j] = (aj == 0);
  }
  __syncthreads();
  const int k = t;
  const float mean = means[k];
  const float sd   = fabsf(stds[k]) + 1e-5f;
  const float inv  = 1.0f / sd;
  float sum = 0.f;
  for (int j = 0; j < Nc; ++j) {
    if (!pad_s[j]) {
      float u = (e_s[j] - mean) * inv;
      sum += expf(-0.5f * u * u);
    }
  }
  ef[(long)row*Kc + k] = sum * inv * 0.39894232f;  // 1/sqrt(2*3.14159)
}

// node[(n*B+b)] = node_tmp[(b*N+n)] + atom_emb[atoms[b,n]]
__global__ __launch_bounds__(256) void add_emb_kernel(
    const float* __restrict__ nt, const int* __restrict__ atoms,
    const float* __restrict__ emb, float* __restrict__ node)
{
  const int row = blockIdx.x;          // b*N + n
  const int b = row / Nc, n = row % Nc;
  const int a = atoms[row];
  const long orow = (long)n * Bc + b;
  for (int d = threadIdx.x; d < Dc; d += 256)
    node[orow*Dc + d] = nt[(long)row*Dc + d] + emb[(long)a*Dc + d];
}

// ---------------------------------------------------------------------------
// Fused attention-bias kernel. For fixed (b,i), rows are j:
//   bias[b,h,i,j] = gelu(gbf(b,i,j,:) @ W1 + b1) @ W2 + b2 ; pad_j -> -1e9
// grid = (N/64, B*N), 256 threads. Two WMMA stages fully in LDS.
// w1/w2 staged TRANSPOSED ([n][k], padded rows) for vector fragment loads.
// ---------------------------------------------------------------------------
__global__ __launch_bounds__(256) void bias_kernel(
    const int* __restrict__ atoms, const float* __restrict__ pos,
    const float* __restrict__ means, const float* __restrict__ stds,
    const float* __restrict__ mul,  const float* __restrict__ gbias,
    const float* __restrict__ w1,   const float* __restrict__ b1,
    const float* __restrict__ w2,   const float* __restrict__ b2,
    float* __restrict__ out)
{
  __shared__ __bf16 tile_s[64][Kc];    // gbf tile, then gelu act (16 KB, row 256B)
  __shared__ __bf16 w1t[Kc][136];      // W1 transposed [n][k], row 272B (34 KB)
  __shared__ __bf16 w2t[16][136];      // W2 transposed [n][k], zero-padded (4.25 KB)
  __shared__ float  e_s[64];
  __shared__ int    pad_s[64];

  const int t  = threadIdx.x;
  const int bi = blockIdx.y;
  const int b  = bi >> 8;              // / 256
  const int i  = bi & 255;
  const int j0 = blockIdx.x * 64;

  if (t < 64) {
    int j  = j0 + t;
    int aj = atoms[b*Nc + j];
    float dx = pos[(b*Nc+i)*3+0] - pos[(b*Nc+j)*3+0];
    float dy = pos[(b*Nc+i)*3+1] - pos[(b*Nc+j)*3+1];
    float dz = pos[(b*Nc+i)*3+2] - pos[(b*Nc+j)*3+2];
    float sq = dx*dx + dy*dy + dz*dz;
    float dist = (sq > 0.f) ? sqrtf(sq) : 0.f;
    int et = atoms[b*Nc+i]*10 + aj;
    e_s[t]   = mul[et]*dist + gbias[et];
    pad_s[t] = (aj == 0);
  }
  // stage W1 transposed: read float4 along n, scatter 4 b16 rows
  #pragma unroll
  for (int it = 0; it < 16; ++it) {    // 4096 float4s / 256 threads
    int item = t + it * 256;           // k = item>>5 (0..127), n4 = (item&31)*4
    int k  = item >> 5;
    int n4 = (item & 31) * 4;
    float4 v = *(const float4*)(w1 + (long)k * Kc + n4);
    w1t[n4+0][k] = (__bf16)v.x; w1t[n4+1][k] = (__bf16)v.y;
    w1t[n4+2][k] = (__bf16)v.z; w1t[n4+3][k] = (__bf16)v.w;
  }
  // stage W2 transposed + pad cols 12..15 with zero
  #pragma unroll
  for (int it = 0; it < 8; ++it) {     // 2048 elements
    int item = t + it * 256;           // n = item>>7, k = item&127
    int n = item >> 7, k = item & 127;
    w2t[n][k] = (__bf16)((n < Hc) ? w2[(long)k * Hc + n] : 0.0f);
  }
  __syncthreads();

  // gbf tile, two k's per store (b32 LDS stores)
  #pragma unroll
  for (int it = 0; it < 16; ++it) {    // 4096 pairs
    int idx = t + it * 256;
    int jl = idx >> 6;                 // 0..63
    int k2 = (idx & 63) * 2;           // even k
    float e = e_s[jl];
    float sd0 = fabsf(stds[k2])   + 1e-5f;
    float sd1 = fabsf(stds[k2+1]) + 1e-5f;
    float i0 = 1.0f / sd0, i1 = 1.0f / sd1;
    float u0 = (e - means[k2])   * i0;
    float u1 = (e - means[k2+1]) * i1;
    v2bf p;
    p[0] = (__bf16)(expf(-0.5f*u0*u0) * i0 * 0.39894232f);
    p[1] = (__bf16)(expf(-0.5f*u1*u1) * i1 * 0.39894232f);
    *(v2bf*)&tile_s[jl][k2] = p;
  }
  __syncthreads();

  const int wave = t >> 5, lane = t & 31;
  const int wm = wave & 3, wn = wave >> 2;
  const int m0 = wm * 16;
  const int mof = (lane >= 16) ? 8 : 0;
  const int hiA = (lane >= 16) ? 8 : 0;
  const int hiB = (lane >= 16) ? 8 : 0;

  // Stage 1: [64,128] @ W1[128,128]
  v8f acc[4];
  #pragma unroll
  for (int s = 0; s < 4; ++s) acc[s] = zero8();
  #pragma unroll
  for (int kk = 0; kk < 4; ++kk) {
    const int k0 = kk * 32;
    const int mr = m0 + (lane & 15);
    v16bf af = cat16(*(const v8bf*)&tile_s[mr][k0 + hiA],
                     *(const v8bf*)&tile_s[mr][k0 + 16 + hiA]);
    #pragma unroll
    for (int s = 0; s < 4; ++s) {
      const int nr = wn * 64 + s * 16 + (lane & 15);
      v16bf bf = cat16(*(const v8bf*)&w1t[nr][k0 + hiB],
                       *(const v8bf*)&w1t[nr][k0 + 16 + hiB]);
      acc[s] = __builtin_amdgcn_wmma_f32_16x16x32_bf16(false, af, false, bf,
                                                       (short)0, acc[s], false, false);
    }
  }
  __syncthreads();                     // everyone done reading tile_s

  // gelu + write activation back into tile_s
  #pragma unroll
  for (int s = 0; s < 4; ++s) {
    int n = wn * 64 + s * 16 + (lane & 15);
    #pragma unroll
    for (int g = 0; g < 8; ++g)
      tile_s[m0 + mof + g][n] = (__bf16)gelu_exact(acc[s][g] + b1[n]);
  }
  __syncthreads();

  // Stage 2: [64,128] @ W2[128,16(12)] on waves 0..3 (wave-uniform branch)
  if (wave < 4) {
    const int mm0 = wave * 16;
    const int n = lane & 15;
    v8f a2 = zero8();
    #pragma unroll
    for (int kk = 0; kk < 4; ++kk) {
      const int k0 = kk * 32;
      const int mr = mm0 + (lane & 15);
      v16bf af = cat16(*(const v8bf*)&tile_s[mr][k0 + hiA],
                       *(const v8bf*)&tile_s[mr][k0 + 16 + hiA]);
      v16bf bf = cat16(*(const v8bf*)&w2t[n][k0 + hiB],
                       *(const v8bf*)&w2t[n][k0 + 16 + hiB]);
      a2 = __builtin_amdgcn_wmma_f32_16x16x32_bf16(false, af, false, bf,
                                                   (short)0, a2, false, false);
    }
    if (n < Hc) {
      #pragma unroll
      for (int g = 0; g < 8; ++g) {
        int m = mm0 + mof + g;
        int j = j0 + m;
        float v = a2[g] + b2[n];
        if (pad_s[m]) v = NEGB;
        out[(((long)(b*Hc + n))*Nc + i)*Nc + j] = v;
      }
    }
  }
}

// ---------------------------------------------------------------------------
// LayerNorm over D=768 per row; swap=1 remaps [N,B] row -> [B,N] output row
// ---------------------------------------------------------------------------
__global__ __launch_bounds__(256) void ln_kernel(
    const float* __restrict__ X, const float* __restrict__ sc,
    const float* __restrict__ bs, float* __restrict__ Y, int swap)
{
  __shared__ float red[256];
  const int row = blockIdx.x, t = threadIdx.x;
  const float* x = X + (long)row * Dc;
  float a0 = x[t], a1 = x[t+256], a2 = x[t+512];
  red[t] = a0 + a1 + a2;
  __syncthreads();
  for (int off = 128; off; off >>= 1) { if (t < off) red[t] += red[t+off]; __syncthreads(); }
  float mean = red[0] * (1.0f/768.0f);
  __syncthreads();
  float d0 = a0-mean, d1 = a1-mean, d2 = a2-mean;
  red[t] = d0*d0 + d1*d1 + d2*d2;
  __syncthreads();
  for (int off = 128; off; off >>= 1) { if (t < off) red[t] += red[t+off]; __syncthreads(); }
  float inv = rsqrtf(red[0] * (1.0f/768.0f) + 1e-5f);
  long orow = row;
  if (swap) { int bb = row % Bc, ii = row / Bc; orow = (long)bb * Nc + ii; }
  float* y = Y + orow * Dc;
  y[t]     = d0*inv*sc[t]     + bs[t];
  y[t+256] = d1*inv*sc[t+256] + bs[t+256];
  y[t+512] = d2*inv*sc[t+512] + bs[t+512];
}

// row softmax over 256 columns, in place
__global__ __launch_bounds__(256) void softmax_kernel(float* __restrict__ S)
{
  __shared__ float red[256];
  const long row = blockIdx.x;
  float* p = S + row * Nc;
  const int t = threadIdx.x;
  float v = p[t];
  red[t] = v; __syncthreads();
  for (int off = 128; off; off >>= 1) { if (t < off) red[t] = fmaxf(red[t], red[t+off]); __syncthreads(); }
  float m = red[0]; __syncthreads();
  float e = expf(v - m);
  red[t] = e; __syncthreads();
  for (int off = 128; off; off >>= 1) { if (t < off) red[t] += red[t+off]; __syncthreads(); }
  p[t] = e / red[0];
}

// ---------------------------------------------------------------------------
extern "C" void kernel_launch(void* const* d_in, const int* in_sizes, int n_in,
                              void* d_out, int out_size, void* d_ws, size_t ws_size,
                              hipStream_t stream) {
  const int*   atoms    = (const int*)  d_in[0];
  const float* pos      = (const float*)d_in[1];
  const float* atom_emb = (const float*)d_in[2];
  const float* gmeans   = (const float*)d_in[3];
  const float* gstds    = (const float*)d_in[4];
  const float* gmul     = (const float*)d_in[5];
  const float* gbias    = (const float*)d_in[6];
  const float* bp_w1    = (const float*)d_in[7];
  const float* bp_b1    = (const float*)d_in[8];
  const float* bp_w2    = (const float*)d_in[9];
  const float* bp_b2    = (const float*)d_in[10];
  const float* edge_w   = (const float*)d_in[11];
  const float* edge_b   = (const float*)d_in[12];
  const float* ap_w     = (const float*)d_in[13];
  const float* ap_b     = (const float*)d_in[14];
  const float* ln1_s    = (const float*)d_in[15];
  const float* ln1_b    = (const float*)d_in[16];
  const float* wi       = (const float*)d_in[17];
  const float* bi       = (const float*)d_in[18];
  const float* wo       = (const float*)d_in[19];
  const float* bo       = (const float*)d_in[20];
  const float* ln2_s    = (const float*)d_in[21];
  const float* ln2_b    = (const float*)d_in[22];
  const float* w1       = (const float*)d_in[23];
  const float* b1       = (const float*)d_in[24];
  const float* w2       = (const float*)d_in[25];
  const float* b2       = (const float*)d_in[26];
  const float* fln_s    = (const float*)d_in[27];
  const float* fln_b    = (const float*)d_in[28];
  float* out = (float*)d_out;
  (void)in_sizes; (void)n_in; (void)out_size; (void)ws_size;

  char* w = (char*)d_ws;
  auto alloc = [&](size_t n) { float* p = (float*)w; w += n * sizeof(float); return p; };
  float* ef      = alloc((size_t)ROWS * Kc);
  float* ntmp    = alloc((size_t)ROWS * Dc);
  float* node    = alloc((size_t)ROWS * Dc);
  float* x       = alloc((size_t)ROWS * Dc);
  float* h       = alloc((size_t)ROWS * Dc);
  float* qkv     = alloc((size_t)ROWS * 3 * Dc);
  float* scores  = alloc((size_t)Bc * Hc * Nc * Nc);
  float* biasb   = alloc((size_t)Bc * Hc * Nc * Nc);
  float* o       = alloc((size_t)ROWS * Dc);
  float* tmp     = alloc((size_t)ROWS * Dc);

  auto gemm = [&](const float* A, long lda, long sA1, long sA2,
                  const float* Bm, long ldb, long sB1, long sB2, int opB,
                  const float* bias_, const float* R, long ldr, long sR1, long sR2,
                  float* C, long ldc, long sC1, long sC2,
                  int M, int Ncols, int K, int zdiv, int nz, float alpha, int gelu) {
    dim3 g(Ncols / 64, M / 64, nz);
    if (opB)
      gemm_bf16_wmma<1><<<g, 256, 0, stream>>>(A, lda, sA1, sA2, Bm, ldb, sB1, sB2,
                                               bias_, R, ldr, sR1, sR2, C, ldc, sC1, sC2,
                                               zdiv, K, alpha, gelu);
    else
      gemm_bf16_wmma<0><<<g, 256, 0, stream>>>(A, lda, sA1, sA2, Bm, ldb, sB1, sB2,
                                               bias_, R, ldr, sR1, sR2, C, ldc, sC1, sC2,
                                               zdiv, K, alpha, gelu);
  };

  // --- preprocessing ---
  ef_kernel<<<ROWS, 128, 0, stream>>>(atoms, pos, gmeans, gstds, gmul, gbias, ef);
  // node_tmp = ef @ edge_w + edge_b               [2048,128]x[128,768]
  gemm(ef, Kc, 0, 0, edge_w, Dc, 0, 0, 0, edge_b, nullptr, 0, 0, 0,
       ntmp, Dc, 0, 0, ROWS, Dc, Kc, 1, 1, 1.0f, 0);
  add_emb_kernel<<<ROWS, 256, 0, stream>>>(ntmp, atoms, atom_emb, node);
  // x = node @ ap_w + ap_b                        [2048,768]x[768,768]
  gemm(node, Dc, 0, 0, ap_w, Dc, 0, 0, 0, ap_b, nullptr, 0, 0, 0,
       x, Dc, 0, 0, ROWS, Dc, Dc, 1, 1, 1.0f, 0);
  // attention bias tensor [B,H,N,N]
  bias_kernel<<<dim3(Nc/64, Bc*Nc), 256, 0, stream>>>(
      atoms, pos, gmeans, gstds, gmul, gbias, bp_w1, bp_b1, bp_w2, bp_b2, biasb);

  const long NN = (long)Nc * Nc;
  const long qrow = (long)Bc * 3 * Dc;   // 18432: row stride of qkv in i
  // --- transformer layers ---
  for (int it = 0; it < 8; ++it) {
    int l = it & 3;
    const float* Wi = wi + (long)l * Dc * 3 * Dc;
    const float* Bi = bi + (long)l * 3 * Dc;
    const float* Wo = wo + (long)l * Dc * Dc;
    const float* Bo = bo + (long)l * Dc;
    const float* W1 = w1 + (long)l * Dc * Dc;
    const float* B1 = b1 + (long)l * Dc;
    const float* W2 = w2 + (long)l * Dc * Dc;
    const float* B2 = b2 + (long)l * Dc;

    ln_kernel<<<ROWS, 256, 0, stream>>>(x, ln1_s + l*Dc, ln1_b + l*Dc, h, 0);
    // qkv = h @ Wi + Bi                           [2048,768]x[768,2304]
    gemm(h, Dc, 0, 0, Wi, 3*Dc, 0, 0, 0, Bi, nullptr, 0, 0, 0,
         qkv, 3*Dc, 0, 0, ROWS, 3*Dc, Dc, 1, 1, 1.0f, 0);
    // scores[b,h] = 0.125 * q @ k^T + bias        batched over z=b*H+h
    gemm(qkv,        qrow, 3*Dc, HDc,              // A = q
         qkv + Dc,   qrow, 3*Dc, HDc, 1,           // B = k (transposed)
         nullptr, biasb, Nc, Hc*NN, NN,
         scores, Nc, Hc*NN, NN,
         Nc, Nc, HDc, Hc, Bc*Hc, 0.125f, 0);
    softmax_kernel<<<Bc*Hc*Nc, 256, 0, stream>>>(scores);
    // o[b,h] = P @ v                              batched over z=b*H+h
    gemm(scores, Nc, Hc*NN, NN,
         qkv + 2*Dc, qrow, 3*Dc, HDc, 0,           // B = v
         nullptr, nullptr, 0, 0, 0,
         o, (long)Bc*Dc, Dc, HDc,
         Nc, HDc, Nc, Hc, Bc*Hc, 1.0f, 0);
    // x = x + o @ Wo + Bo
    gemm(o, Dc, 0, 0, Wo, Dc, 0, 0, 0, Bo, x, Dc, 0, 0,
         x, Dc, 0, 0, ROWS, Dc, Dc, 1, 1, 1.0f, 0);
    ln_kernel<<<ROWS, 256, 0, stream>>>(x, ln2_s + l*Dc, ln2_b + l*Dc, h, 0);
    // tmp = gelu(h @ W1 + B1)
    gemm(h, Dc, 0, 0, W1, Dc, 0, 0, 0, B1, nullptr, 0, 0, 0,
         tmp, Dc, 0, 0, ROWS, Dc, Dc, 1, 1, 1.0f, 1);
    // x = x + tmp @ W2 + B2
    gemm(tmp, Dc, 0, 0, W2, Dc, 0, 0, 0, B2, x, Dc, 0, 0,
         x, Dc, 0, 0, ROWS, Dc, Dc, 1, 1, 1.0f, 0);
  }
  // final LN, writing [B,N,D]
  ln_kernel<<<ROWS, 256, 0, stream>>>(x, fln_s, fln_b, out, 1);
}
